// HeadV1_18468359373428
// MI455X (gfx1250) — matrix operand order repaired and checked by reading, not compiled, not run on hardware
//
#include <hip/hip_runtime.h>
#include <hip/hip_bf16.h>
#include <cstddef>

// ---------------- problem constants ----------------
constexpr int Tdim = 4096;   // sequence length
constexpr int Cdim = 64;     // head size
constexpr int NB   = 4;      // batch
constexpr int QB   = 128;    // query rows per block (8 waves x 16)

typedef __attribute__((ext_vector_type(16))) _Float16 v16h;
typedef __attribute__((ext_vector_type(8)))  _Float16 v8h;
typedef __attribute__((ext_vector_type(8)))  float    v8f;
typedef __attribute__((ext_vector_type(4)))  float    v4f;

static __device__ __forceinline__ v8f wmma_f16(v16h a, v16h b, v8f c) {
  // D = A(16x32 f16) * B(32x16 f16) + C(16x16 f32)
  return __builtin_amdgcn_wmma_f32_16x16x32_f16(
      /*neg_a=*/false, a, /*neg_b=*/false, b,
      /*c_mod=*/(short)0, c, /*reuse_a=*/false, /*reuse_b=*/false);
}

// async copy of 16 bytes global -> LDS, tracked by ASYNCcnt (per-lane)
static __device__ __forceinline__ void async_g2l_b128(const void* gptr,
                                                      void* lptr) {
  const unsigned lds_off = (unsigned)(size_t)lptr;  // low 32b of generic LDS
  asm volatile("global_load_async_to_lds_b128 %0, %1, off"
               :: "v"(lds_off), "v"(gptr) : "memory");
}

// =====================================================================
// Kernel 1: QKV projection.  One wave per 16-row tile of x[B*T, 64].
//   Q (scaled by C^-0.5) -> f16 row-major [B*T][64]
//   K                    -> f16 row-major [B*T][64]
//   V                    -> f16 TRANSPOSED [b][col][T]
// =====================================================================
__global__ __launch_bounds__(256) void qkv_kernel(
    const float* __restrict__ x,  const float* __restrict__ Wk,
    const float* __restrict__ Wq, const float* __restrict__ Wv,
    _Float16* __restrict__ Qh, _Float16* __restrict__ Kh,
    _Float16* __restrict__ Vt)
{
  const int lane = threadIdx.x & 31;
  const int wid  = threadIdx.x >> 5;
  const int tile = blockIdx.x * 8 + wid;      // 16-row tile over B*T
  const int m0g  = tile * 16;                 // global row (batch-contig)
  const int col  = lane & 15;
  const int hi   = lane >> 4;                 // lane half: 0 or 1

  // ---- A fragments of x: lane holds row (lane&15); element e maps to
  // K = 16*hi + e within each 32-wide chunk.
  const float* xr = x + (size_t)(m0g + col) * Cdim;
  v16h a[2];
#pragma unroll
  for (int ch = 0; ch < 2; ++ch) {
    const int c0 = ch * 32 + hi * 16;
    v4f f0 = *(const v4f*)(xr + c0);
    v4f f1 = *(const v4f*)(xr + c0 + 4);
    v4f f2 = *(const v4f*)(xr + c0 + 8);
    v4f f3 = *(const v4f*)(xr + c0 + 12);
#pragma unroll
    for (int e = 0; e < 4; ++e) {
      a[ch][e]      = (_Float16)f0[e];
      a[ch][e + 4]  = (_Float16)f1[e];
      a[ch][e + 8]  = (_Float16)f2[e];
      a[ch][e + 12] = (_Float16)f3[e];
    }
  }

  const float* Wmats[3] = {Wq, Wk, Wv};
  const int bvb   = hi * 16;                  // B-frag K base per lane half
  const int bq    = m0g >> 12;                // batch index (T = 4096)
  const int trow0 = m0g & (Tdim - 1);         // row within batch

#pragma unroll
  for (int w = 0; w < 3; ++w) {
    const float* W = Wmats[w];
#pragma unroll
    for (int nt = 0; nt < 4; ++nt) {
      const int n = nt * 16 + col;            // output column
      // y = x @ W^T  =>  B[k][n] = W[n][k]  (contiguous row read of W)
      const float* wr = W + (size_t)n * Cdim;
      v16h b0, b1;
#pragma unroll
      for (int e = 0; e < 16; ++e) {
        b0[e] = (_Float16)wr[bvb + e];        // K-chunk 0: c in [0,32)
        b1[e] = (_Float16)wr[32 + bvb + e];   // K-chunk 1: c in [32,64)
      }
      v8f acc;
#pragma unroll
      for (int r = 0; r < 8; ++r) acc[r] = 0.f;
      acc = wmma_f16(a[0], b0, acc);
      acc = wmma_f16(a[1], b1, acc);

      // C/D layout: acc[r] holds (row = r + 8*hi, col = lane&15)
      if (w == 0) {                           // Q, pre-scaled by 1/sqrt(C)
#pragma unroll
        for (int r = 0; r < 8; ++r) {
          const int row = m0g + r + hi * 8;
          Qh[(size_t)row * Cdim + n] = (_Float16)(acc[r] * 0.125f);
        }
      } else if (w == 1) {                    // K row-major
#pragma unroll
        for (int r = 0; r < 8; ++r) {
          const int row = m0g + r + hi * 8;
          Kh[(size_t)row * Cdim + n] = (_Float16)acc[r];
        }
      } else {                                // V transposed [b][col][T]
#pragma unroll
        for (int r = 0; r < 8; ++r) {
          const int trow = trow0 + r + hi * 8;
          Vt[((size_t)(bq * Cdim + n)) * Tdim + trow] = (_Float16)acc[r];
        }
      }
    }
  }
}

// =====================================================================
// Kernel 2: causal flash attention.
//   Block = 8 waves covering 128 consecutive query rows of one batch.
//   Key blocks of 32; K/V tiles staged cooperatively into LDS with
//   double-buffered global_load_async_to_lds_b128 (ASYNCcnt pipeline);
//   all 8 waves consume the same LDS tile.
// =====================================================================
__global__ __launch_bounds__(256) void attn_kernel(
    const _Float16* __restrict__ Qh, const _Float16* __restrict__ Kh,
    const _Float16* __restrict__ Vt, float* __restrict__ out)
{
  __shared__ _Float16 KtileS[2][32 * Cdim];   // [key][c]   2 x 4 KB
  __shared__ _Float16 VtileS[2][Cdim * 32];   // [col][key] 2 x 4 KB
  __shared__ _Float16 Pbuf[8][16 * 32];       // per-wave P staging, 8 KB

  const int tid  = threadIdx.x;
  const int lane = tid & 31;
  const int wid  = tid >> 5;
  const int b    = blockIdx.x >> 5;           // 32 blocks per batch
  const int q0   = (blockIdx.x & 31) * QB;    // block query base (in batch)
  const int m0   = q0 + wid * 16;             // this wave's query tile
  const int col  = lane & 15;
  const int hi   = lane >> 4;

  const _Float16* Ksrc = Kh + (size_t)b * Tdim * Cdim;
  const _Float16* Vsrc = Vt + (size_t)b * Cdim * Tdim;

  // staging slices: 256 threads x 16B = 4 KB per tile
  const int krow = tid >> 3;                  // key row 0..31
  const int kcol = (tid & 7) * 8;             // 8 halves of the 64-wide row
  const int vrow = tid >> 2;                  // v column 0..63
  const int vkh  = (tid & 3) * 8;             // 8 halves of the 32-key row

  // ---- Q A-fragments (already scaled by C^-0.5) ----
  const _Float16* qr = Qh + (size_t)(b * Tdim + m0 + col) * Cdim;
  v16h qa[2];
#pragma unroll
  for (int ch = 0; ch < 2; ++ch) {
    const int c0 = ch * 32 + hi * 16;
    v8h lo = *(const v8h*)(qr + c0);
    v8h hh = *(const v8h*)(qr + c0 + 8);
#pragma unroll
    for (int e = 0; e < 8; ++e) { qa[ch][e] = lo[e]; qa[ch][e + 8] = hh[e]; }
  }

  v8f acc[4];                                 // 16x64 output accumulator
#pragma unroll
  for (int nt = 0; nt < 4; ++nt)
#pragma unroll
    for (int r = 0; r < 8; ++r) acc[nt][r] = 0.f;

  float m_i[8], l_i[8];
#pragma unroll
  for (int r = 0; r < 8; ++r) { m_i[r] = -3.0e38f; l_i[r] = 0.f; }

  const int kend  = m0 + 16;                  // this wave's causal bound
  const int niter = (q0 + QB) / 32;           // block-wide key blocks
  _Float16* pb = &Pbuf[wid][0];

  // prologue: stage tile 0 into buffer 0
  async_g2l_b128(Ksrc + (size_t)krow * Cdim + kcol,
                 &KtileS[0][krow * Cdim + kcol]);
  async_g2l_b128(Vsrc + (size_t)vrow * Tdim + vkh,
                 &VtileS[0][vrow * 32 + vkh]);

  for (int it = 0; it < niter; ++it) {
    const int k0  = it * 32;
    const int buf = it & 1;

    if (it + 1 < niter) {                     // stream next tile into other buf
      const int kn = k0 + 32;
      async_g2l_b128(Ksrc + (size_t)(kn + krow) * Cdim + kcol,
                     &KtileS[buf ^ 1][krow * Cdim + kcol]);
      async_g2l_b128(Vsrc + (size_t)vrow * Tdim + kn + vkh,
                     &VtileS[buf ^ 1][vrow * 32 + vkh]);
      // async loads complete in order: <=2 outstanding => tile `it` landed
      asm volatile("s_wait_asynccnt 0x2" ::: "memory");
    } else {
      asm volatile("s_wait_asynccnt 0x0" ::: "memory");
    }
    __syncthreads();                          // tile `it` visible to all waves

    if (k0 < kend) {                          // causal: skip fully-masked blocks
      // ---- S = Q @ K^T for 32 keys (two 16-key subtiles) ----
      v8f s[2];
#pragma unroll
      for (int sub = 0; sub < 2; ++sub) {
        const _Float16* kr = &KtileS[buf][(sub * 16 + col) * Cdim + hi * 16];
        v16h bk0 = *(const v16h*)(kr);
        v16h bk1 = *(const v16h*)(kr + 32);
        v8f z;
#pragma unroll
        for (int r = 0; r < 8; ++r) z[r] = 0.f;
        z = wmma_f16(qa[0], bk0, z);
        z = wmma_f16(qa[1], bk1, z);
        s[sub] = z;
      }

      // ---- online softmax (rows live in VGPR index + lane-half) ----
      float alpha[8];
#pragma unroll
      for (int r = 0; r < 8; ++r) {
        const int qg = m0 + r + hi * 8;
        float v0 = (k0 + col      <= qg) ? s[0][r] : -3.0e38f;
        float v1 = (k0 + 16 + col <= qg) ? s[1][r] : -3.0e38f;
        float mx = fmaxf(v0, v1);
#pragma unroll
        for (int off = 1; off < 16; off <<= 1)  // stays within 16-lane half
          mx = fmaxf(mx, __shfl_xor(mx, off, 32));
        const float mn = fmaxf(m_i[r], mx);
        const float al = __expf(m_i[r] - mn);
        const float p0 = __expf(v0 - mn);
        const float p1 = __expf(v1 - mn);
        float rs = p0 + p1;
#pragma unroll
        for (int off = 1; off < 16; off <<= 1)
          rs += __shfl_xor(rs, off, 32);
        l_i[r]   = l_i[r] * al + rs;
        m_i[r]   = mn;
        alpha[r] = al;
        // P (C-layout) -> LDS row-major [16][32] f16
        pb[(r + hi * 8) * 32 + col]      = (_Float16)p0;
        pb[(r + hi * 8) * 32 + 16 + col] = (_Float16)p1;
      }
#pragma unroll
      for (int nt = 0; nt < 4; ++nt)
#pragma unroll
        for (int r = 0; r < 8; ++r) acc[nt][r] *= alpha[r];

      // intra-wave LDS RAW fence before re-reading P in A-layout
      asm volatile("s_wait_dscnt 0x0" ::: "memory");

      // ---- P A-fragment (16x32 f16) from LDS ----
      v16h pa;
      {
        const _Float16* pr = pb + col * 32 + hi * 16;
        v8h lo = *(const v8h*)(pr);
        v8h hh = *(const v8h*)(pr + 8);
#pragma unroll
        for (int e = 0; e < 8; ++e) { pa[e] = lo[e]; pa[e + 8] = hh[e]; }
      }

      // ---- O += P @ V (V tile is [col][key] in LDS: contiguous 32B) ----
#pragma unroll
      for (int nt = 0; nt < 4; ++nt) {
        const _Float16* vr = &VtileS[buf][(nt * 16 + col) * 32 + hi * 16];
        v16h bv = *(const v16h*)vr;
        acc[nt] = wmma_f16(pa, bv, acc[nt]);
      }
    }

    __syncthreads();   // all waves done with buf before it is restaged
  }

  // ---- normalize and write out (f32) ----
#pragma unroll
  for (int nt = 0; nt < 4; ++nt) {
#pragma unroll
    for (int r = 0; r < 8; ++r) {
      const int row = m0 + r + hi * 8;
      out[((size_t)(b * Tdim + row)) * Cdim + nt * 16 + col] =
          acc[nt][r] / l_i[r];
    }
  }
}

// =====================================================================
extern "C" void kernel_launch(void* const* d_in, const int* in_sizes, int n_in,
                              void* d_out, int out_size, void* d_ws, size_t ws_size,
                              hipStream_t stream) {
  (void)in_sizes; (void)n_in; (void)out_size; (void)ws_size;
  const float* x  = (const float*)d_in[0];
  const float* Wk = (const float*)d_in[1];
  const float* Wq = (const float*)d_in[2];
  const float* Wv = (const float*)d_in[3];
  float* out = (float*)d_out;

  // workspace: Qh | Kh | Vt, each B*T*C f16 = 2 MB (6 MB total)
  const size_t elems = (size_t)NB * Tdim * Cdim;
  char* ws = (char*)d_ws;
  _Float16* Qh = (_Float16*)(ws);
  _Float16* Kh = (_Float16*)(ws + 2 * elems);
  _Float16* Vt = (_Float16*)(ws + 4 * elems);

  const int tiles = NB * Tdim / 16;           // 1024 tiles, 8 waves/block
  dim3 grid(tiles / 8);                       // 128 blocks x 256 threads
  qkv_kernel<<<grid, 256, 0, stream>>>(x, Wk, Wq, Wv, Qh, Kh, Vt);
  attn_kernel<<<grid, 256, 0, stream>>>(Qh, Kh, Vt, out);
}
